// VectorQuantizer_58849641890161
// MI455X (gfx1250) — matrix operand order repaired and checked by reading, not compiled, not run on hardware
//
#include <hip/hip_runtime.h>
#include <hip/hip_bf16.h>

// ---------------------------------------------------------------------------
// VQ on MI455X (gfx1250, wave32):
//   indices[b] = argmin_n ( e2[n] - 2 * dot(x_b, c_n) )      (x^2, sqrt dropped)
//   quantized_st = x + (c[idx] - x) ; vq_loss = 1.5 * mean((c[idx]-x)^2)
// GEMM done with v_wmma_f32_16x16x32_bf16 using a hi/lo bf16 split of fp32
// (3 WMMAs per k-step: hi*hi + hi*lo + lo*hi) for ~fp32-accurate scores.
// ---------------------------------------------------------------------------

typedef __attribute__((ext_vector_type(16))) __bf16 v16bf;
typedef __attribute__((ext_vector_type(8)))  float  v8f;

constexpr int Bn = 32768;   // batch rows
constexpr int Kn = 8192;    // codebook entries
constexpr int Dn = 256;     // feature dim (WMMA contraction dim)
constexpr int TILE_ROWS = 128;   // x rows per workgroup (8 waves x 16)
constexpr int CHUNK = 32;        // codebook rows per K-iteration (2 tiles/wave)

__device__ __forceinline__ unsigned short f2bf(float f) {
    unsigned int u = __float_as_uint(f);
    u += 0x7FFFu + ((u >> 16) & 1u);          // round-to-nearest-even
    return (unsigned short)(u >> 16);
}
__device__ __forceinline__ float bf2f(unsigned short h) {
    return __uint_as_float(((unsigned int)h) << 16);
}

// Load a 16-half WMMA fragment from LDS as two 16-byte chunks.
__device__ __forceinline__ v16bf load_frag(const unsigned short* p, int off0, int off1) {
    union { uint4 u[2]; v16bf v; } f;
    f.u[0] = *reinterpret_cast<const uint4*>(p + off0);
    f.u[1] = *reinterpret_cast<const uint4*>(p + off1);
    return f.v;
}

__device__ __forceinline__ v8f wmma_bf16(v16bf a, v16bf b, v8f c) {
    return __builtin_amdgcn_wmma_f32_16x16x32_bf16(
        /*neg_a=*/false, a, /*neg_b=*/false, b,
        /*c_mod=*/(short)0, c, /*reuse_a=*/false, /*reuse_b=*/false);
}

// ---------------------------------------------------------------------------
// Kernel 1: codebook row norms e2[k] = ||c_k||^2 (fp32 exact)
// ---------------------------------------------------------------------------
__global__ __launch_bounds__(256) void vq_e2_kernel(const float* __restrict__ cb,
                                                    float* __restrict__ e2) {
    int k = blockIdx.x * 256 + threadIdx.x;
    if (k >= Kn) return;
    const float4* r4 = reinterpret_cast<const float4*>(cb + (size_t)k * Dn);
    float s = 0.f;
    #pragma unroll 4
    for (int i = 0; i < Dn / 4; ++i) {
        float4 v = r4[i];
        s += v.x * v.x + v.y * v.y + v.z * v.z + v.w * v.w;
    }
    e2[k] = s;
}

// ---------------------------------------------------------------------------
// Kernel 2: fused GEMM + running argmin.
// WG = 256 threads (8 waves). Wave w owns x rows [wgBase + 16w, +16).
// LDS: x tile hi/lo bf16 (128x256 each), codebook chunk hi/lo (32x256 each).
// ---------------------------------------------------------------------------
__global__ __launch_bounds__(256) void vq_argmin_kernel(
    const float* __restrict__ x, const float* __restrict__ cb,
    const float* __restrict__ e2, int* __restrict__ idxOut,
    float* __restrict__ idxOutF) {

    __shared__ __align__(16) unsigned short s_xh[TILE_ROWS * Dn];  // 64 KB
    __shared__ __align__(16) unsigned short s_xl[TILE_ROWS * Dn];  // 64 KB
    __shared__ __align__(16) unsigned short s_ch[CHUNK * Dn];      // 16 KB
    __shared__ __align__(16) unsigned short s_cl[CHUNK * Dn];      // 16 KB

    const int tid  = threadIdx.x;
    const int lane = tid & 31;
    const int wv   = tid >> 5;
    const int ln   = lane & 15;
    const int rowBase = blockIdx.x * TILE_ROWS;

    // ---- load + split x tile into LDS (hi/lo bf16) ----
    {
        const float4* xg4 = reinterpret_cast<const float4*>(x + (size_t)rowBase * Dn);
        for (int t = tid; t < TILE_ROWS * Dn / 4; t += 256) {
            float4 v = xg4[t];
            unsigned short h0 = f2bf(v.x), h1 = f2bf(v.y), h2 = f2bf(v.z), h3 = f2bf(v.w);
            unsigned short l0 = f2bf(v.x - bf2f(h0)), l1 = f2bf(v.y - bf2f(h1));
            unsigned short l2 = f2bf(v.z - bf2f(h2)), l3 = f2bf(v.w - bf2f(h3));
            int hp = t * 4;
            *reinterpret_cast<uint2*>(s_xh + hp) =
                make_uint2((unsigned)h0 | ((unsigned)h1 << 16), (unsigned)h2 | ((unsigned)h3 << 16));
            *reinterpret_cast<uint2*>(s_xl + hp) =
                make_uint2((unsigned)l0 | ((unsigned)l1 << 16), (unsigned)l2 | ((unsigned)l3 << 16));
        }
    }

    float mval[8];
    int   midx[8];
    #pragma unroll
    for (int r = 0; r < 8; ++r) { mval[r] = 3.4e38f; midx[r] = 0; }

    const v8f vzero = {0.f, 0.f, 0.f, 0.f, 0.f, 0.f, 0.f, 0.f};
    const int aRow = wv * 16 + ln;                 // A-tile row for this lane
    const int aK   = (lane >= 16) ? 16 : 0;        // A fragment K sub-offset
    const int bK   = (lane >= 16) ? 16 : 0;        // B fragment K sub-offset
    const float4* cbg4 = reinterpret_cast<const float4*>(cb);

    for (int c0 = 0; c0 < Kn; c0 += CHUNK) {
        __syncthreads();
        // ---- load + split codebook chunk (rows c0..c0+31) ----
        int base4 = (c0 * Dn) / 4;
        for (int t = tid; t < CHUNK * Dn / 4; t += 256) {
            float4 v = cbg4[base4 + t];
            unsigned short h0 = f2bf(v.x), h1 = f2bf(v.y), h2 = f2bf(v.z), h3 = f2bf(v.w);
            unsigned short l0 = f2bf(v.x - bf2f(h0)), l1 = f2bf(v.y - bf2f(h1));
            unsigned short l2 = f2bf(v.z - bf2f(h2)), l3 = f2bf(v.w - bf2f(h3));
            int hp = t * 4;
            *reinterpret_cast<uint2*>(s_ch + hp) =
                make_uint2((unsigned)h0 | ((unsigned)h1 << 16), (unsigned)h2 | ((unsigned)h3 << 16));
            *reinterpret_cast<uint2*>(s_cl + hp) =
                make_uint2((unsigned)l0 | ((unsigned)l1 << 16), (unsigned)l2 | ((unsigned)l3 << 16));
        }
        // prefetch next chunk (256 threads x 128B = 32 KB chunk)
        int nc = c0 + CHUNK;
        if (nc < Kn) __builtin_prefetch(cb + (size_t)nc * Dn + tid * 32, 0, 1);
        __syncthreads();

        v8f acc0 = vzero, acc1 = vzero;
        #pragma unroll
        for (int ks = 0; ks < Dn / 32; ++ks) {
            const int kk = ks * 32;
            v16bf ah  = load_frag(s_xh + aRow * Dn,        kk + aK, kk + aK + 8);
            v16bf al  = load_frag(s_xl + aRow * Dn,        kk + aK, kk + aK + 8);
            v16bf b0h = load_frag(s_ch + ln * Dn,          kk + bK, kk + bK + 8);
            v16bf b0l = load_frag(s_cl + ln * Dn,          kk + bK, kk + bK + 8);
            v16bf b1h = load_frag(s_ch + (16 + ln) * Dn,   kk + bK, kk + bK + 8);
            v16bf b1l = load_frag(s_cl + (16 + ln) * Dn,   kk + bK, kk + bK + 8);
            acc0 = wmma_bf16(ah, b0h, acc0);
            acc0 = wmma_bf16(ah, b0l, acc0);
            acc0 = wmma_bf16(al, b0h, acc0);
            acc1 = wmma_bf16(ah, b1h, acc1);
            acc1 = wmma_bf16(ah, b1l, acc1);
            acc1 = wmma_bf16(al, b1h, acc1);
        }

        // score[n] = e2[n] - 2*dot ; update running per-lane (min, argmin)
        const float e2a = e2[c0 + ln];
        const float e2b = e2[c0 + 16 + ln];
        const int na = c0 + ln, nb = na + 16;
        #pragma unroll
        for (int r = 0; r < 8; ++r) {
            float s0 = __builtin_fmaf(-2.f, acc0[r], e2a);
            if (s0 < mval[r]) { mval[r] = s0; midx[r] = na; }
            float s1 = __builtin_fmaf(-2.f, acc1[r], e2b);
            if (s1 < mval[r]) { mval[r] = s1; midx[r] = nb; }
        }
    }

    // ---- reduce (min, argmin) across the 16 lanes of each column group ----
    #pragma unroll
    for (int r = 0; r < 8; ++r) {
        float v = mval[r];
        int   ix = midx[r];
        #pragma unroll
        for (int off = 1; off < 16; off <<= 1) {
            float ov = __shfl_xor(v, off, 32);
            int   oi = __shfl_xor(ix, off, 32);
            if (ov < v || (ov == v && oi < ix)) { v = ov; ix = oi; }
        }
        if (ln == 0) {
            // C layout: VGPR r holds row r (lanes 0-15) and row r+8 (lanes 16-31)
            int row = rowBase + wv * 16 + ((lane >> 4) << 3) + r;
            idxOut[row]  = ix;
            idxOutF[row] = (float)ix;
        }
    }
}

// ---------------------------------------------------------------------------
// Kernel 3a: zero the loss accumulator (d_out is poisoned before timing).
// ---------------------------------------------------------------------------
__global__ void vq_zero_loss(float* loss) { if (threadIdx.x == 0) *loss = 0.f; }

// ---------------------------------------------------------------------------
// Kernel 3b: gather codebook rows, straight-through output, fused loss.
// One 64-thread block (2 waves) per row; float4 per thread.
// ---------------------------------------------------------------------------
__global__ __launch_bounds__(64) void vq_gather_loss_kernel(
    const float* __restrict__ x, const float* __restrict__ cb,
    const int* __restrict__ idx, float* __restrict__ outQ,
    float* __restrict__ loss) {
    const int row = blockIdx.x;
    const int t = threadIdx.x;
    const int k = idx[row];
    const float4 q  = reinterpret_cast<const float4*>(cb + (size_t)k   * Dn)[t];
    const float4 xv = reinterpret_cast<const float4*>(x  + (size_t)row * Dn)[t];
    float4 o;
    float p = 0.f, d;
    d = q.x - xv.x; o.x = xv.x + d; p += d * d;
    d = q.y - xv.y; o.y = xv.y + d; p += d * d;
    d = q.z - xv.z; o.z = xv.z + d; p += d * d;
    d = q.w - xv.w; o.w = xv.w + d; p += d * d;
    reinterpret_cast<float4*>(outQ + (size_t)row * Dn)[t] = o;

    for (int off = 16; off; off >>= 1) p += __shfl_down(p, off, 32);
    __shared__ float sred[2];
    if ((t & 31) == 0) sred[t >> 5] = p;
    __syncthreads();
    if (t == 0) {
        // vq_loss = (1 + BETA) * mean((q-x)^2), BETA = 0.5
        const float scale = 1.5f / ((float)Bn * (float)Dn);
        atomicAdd(loss, (sred[0] + sred[1]) * scale);
    }
}

// ---------------------------------------------------------------------------
extern "C" void kernel_launch(void* const* d_in, const int* in_sizes, int n_in,
                              void* d_out, int out_size, void* d_ws, size_t ws_size,
                              hipStream_t stream) {
    const float* x  = (const float*)d_in[0];   // (32768, 256) f32
    const float* cb = (const float*)d_in[1];   // (8192, 256) f32

    float* out      = (float*)d_out;
    float* outQ     = out;                                   // (B, D)
    float* outIdxF  = out + (size_t)Bn * Dn;                 // (B,) as float
    float* outLoss  = out + (size_t)Bn * Dn + Bn;            // scalar

    float* e2   = (float*)d_ws;                              // K floats
    int*   idxW = (int*)((char*)d_ws + (size_t)Kn * sizeof(float)); // B ints

    vq_e2_kernel<<<Kn / 256, 256, 0, stream>>>(cb, e2);
    vq_argmin_kernel<<<Bn / TILE_ROWS, 256, 0, stream>>>(x, cb, e2, idxW, outIdxF);
    vq_zero_loss<<<1, 32, 0, stream>>>(outLoss);
    vq_gather_loss_kernel<<<Bn, 64, 0, stream>>>(x, cb, idxW, outQ, outLoss);
}
